// SparseDownsample_3968549781706
// MI455X (gfx1250) — compile-verified, matrix operand order checked by reading, not run on hardware
//
#include <hip/hip_runtime.h>
#include <hip/hip_bf16.h>

// ---------------------------------------------------------------------------
// Sparse strided conv (kernel=2, stride=2, 3D): out[out_idx[i]] += feats[i] @ W[off[i]]
// feats: [N,64] f32   W: [8,64,128] f32   out: [num_out,128] f32
//
// Plan: counting-sort rows by offset k (16-padded buckets) so each 16-row tile
// shares one W[k]; per tile one wave32 does 16x v_wmma_f32_16x16x32_bf16
// (K=64 in 2 steps, Cout=128 in 8 n-tiles) and scatters via f32 atomics.
// ---------------------------------------------------------------------------

#define C_IN   64
#define C_OUT  128
#define K_VOL  8

typedef __attribute__((ext_vector_type(16))) __bf16          v16bf;
typedef __attribute__((ext_vector_type(16))) unsigned short  v16u;
typedef __attribute__((ext_vector_type(8)))  float           v8f;

__device__ __forceinline__ unsigned short f2bf(float f) {
    // round-to-nearest-even f32 -> bf16
    unsigned u = __builtin_bit_cast(unsigned, f);
    unsigned r = u + 0x7FFFu + ((u >> 16) & 1u);
    return (unsigned short)(r >> 16);
}

// meta layout (ints): [0..8]=starts  [16..23]=cursor  [32..39]=counts  [48]=numTiles
// perm begins at meta+64.

__global__ void k_init(int* __restrict__ meta, int permN) {
    int i = blockIdx.x * blockDim.x + threadIdx.x;
    if (i < 64)    meta[i] = 0;
    if (i < permN) meta[64 + i] = -1;
}

__global__ void k_hist(const int* __restrict__ off, int* __restrict__ counts, int n) {
    int i = blockIdx.x * blockDim.x + threadIdx.x;
    if (i < n) atomicAdd(&counts[off[i]], 1);
}

__global__ void k_scan(int* __restrict__ meta) {
    if (blockIdx.x == 0 && threadIdx.x == 0) {
        int s = 0;
        for (int k = 0; k < K_VOL; ++k) {
            meta[k]      = s;              // starts
            meta[16 + k] = s;              // cursor
            s += (meta[32 + k] + 15) & ~15;
        }
        meta[8]  = s;
        meta[48] = s >> 4;                 // numTiles
    }
}

__global__ void k_scatter(const int* __restrict__ off, int* __restrict__ cursor,
                          int* __restrict__ perm, int n) {
    int i = blockIdx.x * blockDim.x + threadIdx.x;
    if (i < n) {
        int p = atomicAdd(&cursor[off[i]], 1);
        perm[p] = i;
    }
}

// Repack W (f32 [8][64][128]) into bf16 wave32 B-fragment layout:
// Wb[k][ks][nt][lane][e] = bf16( W[k][ks*32 + (lane>>4)*16 + e][nt*16 + (lane&15)] )
__global__ void k_wrepack(const float* __restrict__ W, unsigned short* __restrict__ Wb) {
    int e = blockIdx.x * blockDim.x + threadIdx.x;
    if (e >= K_VOL * 2 * 8 * 32 * 16) return;
    int j    =  e        & 15;
    int lane = (e >> 4)  & 31;
    int nt   = (e >> 9)  & 7;
    int ks   = (e >> 12) & 1;
    int k    = (e >> 13) & 7;
    int K = ks * 32 + ((lane >> 4) << 4) + j;
    int N = nt * 16 + (lane & 15);
    Wb[e] = f2bf(W[((size_t)k * C_IN + K) * C_OUT + N]);
}

__global__ void k_zero(float* __restrict__ out, int n) {
    int i = blockIdx.x * blockDim.x + threadIdx.x;
    if (i < n) out[i] = 0.0f;
}

__global__ __launch_bounds__(128)
void spconv_wmma(const float* __restrict__ feats,
                 const unsigned short* __restrict__ Wb,
                 const int* __restrict__ perm,
                 const int* __restrict__ out_idx,
                 const int* __restrict__ starts,     // meta[0..8]
                 const int* __restrict__ numTilesP,  // meta+48
                 float* __restrict__ out)
{
    const int lane = threadIdx.x & 31;
    const int wave = threadIdx.x >> 5;
    const int t    = blockIdx.x * 4 + wave;          // one wave32 per 16-row tile
    if (t >= *numTilesP) return;                     // wave-uniform: EXEC stays all-1s

    const int base = t << 4;

    // bucket k for this tile (buckets are 16-aligned, so a tile never straddles)
    int k = 0;
#pragma unroll
    for (int i = 1; i < K_VOL; ++i) k = (base >= starts[i]) ? i : k;

    const int m   = lane & 15;                       // A-row this lane feeds
    const int hi  = lane >> 4;                       // lanes 16..31 carry K+16
    const int row = perm[base + m];
    const int rr  = (row < 0) ? 0 : row;

    // Gather A: channels [hi*16, hi*16+16) for ks=0, +32 for ks=1
    const float4* fp4 = (const float4*)(feats + (size_t)rr * C_IN + (hi << 4));
    float bufA[32];
#pragma unroll
    for (int j = 0; j < 4; ++j) {
        float4 a = fp4[j];          // ks = 0
        float4 b = fp4[j + 8];      // ks = 1 (+32 floats)
        bufA[j * 4 + 0]      = a.x; bufA[j * 4 + 1]      = a.y;
        bufA[j * 4 + 2]      = a.z; bufA[j * 4 + 3]      = a.w;
        bufA[16 + j * 4 + 0] = b.x; bufA[16 + j * 4 + 1] = b.y;
        bufA[16 + j * 4 + 2] = b.z; bufA[16 + j * 4 + 3] = b.w;
    }
    v16u au0, au1;
#pragma unroll
    for (int e = 0; e < 16; ++e) {
        au0[e] = (row < 0) ? (unsigned short)0 : f2bf(bufA[e]);
        au1[e] = (row < 0) ? (unsigned short)0 : f2bf(bufA[16 + e]);
    }
    v16bf afrag[2];
    afrag[0] = __builtin_bit_cast(v16bf, au0);
    afrag[1] = __builtin_bit_cast(v16bf, au1);

    v8f acc[8];
    const v8f z = {0.f, 0.f, 0.f, 0.f, 0.f, 0.f, 0.f, 0.f};
#pragma unroll
    for (int nt = 0; nt < 8; ++nt) acc[nt] = z;

    const unsigned short* wbase = Wb + ((size_t)k * 2 * 8 * 32 * 16);
#pragma unroll
    for (int ks = 0; ks < 2; ++ks) {
        v16bf a = afrag[ks];
#pragma unroll
        for (int nt = 0; nt < 8; ++nt) {
            const v16u* bp = (const v16u*)(wbase + (((size_t)ks * 8 + nt) * 32 + lane) * 16);
            v16bf b = __builtin_bit_cast(v16bf, *bp);
            acc[nt] = __builtin_amdgcn_wmma_f32_16x16x32_bf16(
                false, a, false, b, (short)0, acc[nt], false, false);
        }
    }

    // Scatter: C layout — VGPR v holds M = v + 8*hi, N = nt*16 + (lane&15)
#pragma unroll
    for (int v = 0; v < 8; ++v) {
        int prow = perm[base + v + (hi << 3)];
        if (prow >= 0) {
            int orow = out_idx[prow];
            float* orp = out + (size_t)orow * C_OUT + (lane & 15);
#pragma unroll
            for (int nt = 0; nt < 8; ++nt) {
                atomicAdd(orp + nt * 16, acc[nt][v]);
            }
        }
    }
}

extern "C" void kernel_launch(void* const* d_in, const int* in_sizes, int n_in,
                              void* d_out, int out_size, void* d_ws, size_t ws_size,
                              hipStream_t stream) {
    const float* feats = (const float*)d_in[0];
    const float* W     = (const float*)d_in[1];
    const int*   off   = (const int*)d_in[2];
    const int*   oidx  = (const int*)d_in[3];
    (void)n_in; (void)ws_size;

    const int N     = in_sizes[0] / C_IN;
    const int permN = N + K_VOL * 16;          // worst-case 16-padding per bucket

    int* meta = (int*)d_ws;
    int* perm = meta + 64;
    unsigned short* Wb = (unsigned short*)(perm + ((permN + 63) & ~63));
    float* out = (float*)d_out;

    const int T = 256;
    k_init   <<<(permN + T - 1) / T, T, 0, stream>>>(meta, permN);
    k_hist   <<<(N + T - 1) / T,     T, 0, stream>>>(off, meta + 32, N);
    k_scan   <<<1, 1, 0, stream>>>(meta);
    k_scatter<<<(N + T - 1) / T,     T, 0, stream>>>(off, meta + 16, perm, N);
    k_wrepack<<<(K_VOL * 2 * 8 * 32 * 16 + T - 1) / T, T, 0, stream>>>(W, Wb);
    k_zero   <<<(out_size + T - 1) / T, T, 0, stream>>>(out, out_size);

    const int maxTiles = N / 16 + K_VOL + 1;   // >= padded total / 16
    spconv_wmma<<<(maxTiles + 3) / 4, 128, 0, stream>>>(
        feats, Wb, perm, oidx, meta, meta + 48, out);
}
